// Bidirectional_GRU_Encoder_19078244729404
// MI455X (gfx1250) — matrix operand order repaired and checked by reading, not compiled, not run on hardware
//
#include <hip/hip_runtime.h>
#include <hip/hip_bf16.h>
#include <stdint.h>

// Bidirectional 2-layer GRU encoder for MI455X (gfx1250, wave32, WMMA).
// B=64, T=512, I=256, H=512, G=3H=1536.
#define B_  64
#define T_  512
#define I_  256
#define H_  512
#define G_  1536
#define BT_ (B_ * T_)
#define HS_ 520   // padded LDS row stride (bf16 elems): 1040B -> bank-conflict-free rows

typedef __attribute__((ext_vector_type(16))) __bf16 v16bf;
typedef __attribute__((ext_vector_type(8)))  float  v8f;

// ---------- helpers ----------------------------------------------------------

__device__ __forceinline__ unsigned short f2bf(float f) {
  // round-to-nearest-even fp32 -> bf16
  unsigned u   = __float_as_uint(f);
  unsigned rnd = 0x7FFFu + ((u >> 16) & 1u);
  return (unsigned short)((u + rnd) >> 16);
}

// A fragment: 16x32 bf16 tile, row-major source, base points at (row0, k0).
// Lanes 0-15: M=lane, K = 0..7 then 16..23 ; lanes 16-31: M=lane-16, K = 8..15 then 24..31.
__device__ __forceinline__ v16bf load_a_frag(const unsigned short* __restrict__ base,
                                             int lda, int lane) {
  int m  = lane & 15;
  int kb = (lane & 16) ? 8 : 0;
  const unsigned short* p = base + (size_t)m * lda + kb;
  union { v16bf v; uint4 q[2]; } u;
  u.q[0] = *(const uint4*)(p);        // K = kb .. kb+7
  u.q[1] = *(const uint4*)(p + 16);   // K = kb+16 .. kb+23
  return u.v;
}

// Same fragment shape but source lives in LDS -> ds_load_b128 pairs.
__device__ __forceinline__ v16bf load_a_frag_lds(const unsigned short* base,
                                                 int lda, int lane) {
  int m  = lane & 15;
  int kb = (lane & 16) ? 8 : 0;
  const unsigned short* p = base + m * lda + kb;
  union { v16bf v; uint4 q[2]; } u;
  u.q[0] = *(const uint4*)(p);
  u.q[1] = *(const uint4*)(p + 16);
  return u.v;
}

// B fragment: 32x16 bf16 tile where B[k][n] = W[n0+n][k0+k] (W row-major [N,K]).
// Lanes 0-15: N=lane, K = 0..7 then 16..23 ; lanes 16-31: N=lane-16, K = 8..15 then 24..31.
__device__ __forceinline__ v16bf load_b_frag(const unsigned short* __restrict__ wbase,
                                             int ldw, int lane) {
  int n  = lane & 15;
  int kb = (lane & 16) ? 8 : 0;
  const unsigned short* p = wbase + (size_t)n * ldw + kb;
  union { v16bf v; uint4 q[2]; } u;
  u.q[0] = *(const uint4*)(p);        // K = kb .. kb+7
  u.q[1] = *(const uint4*)(p + 16);   // K = kb+16 .. kb+23
  return u.v;
}

// C store: 16x16 f32. VGPR v -> row (lane<16 ? v : v+8), col = lane&15.
__device__ __forceinline__ void store_c(float* __restrict__ cbase, int ldc,
                                        int lane, v8f c) {
  int col  = lane & 15;
  int row0 = (lane & 16) ? 8 : 0;
#pragma unroll
  for (int v = 0; v < 8; ++v)
    cbase[(size_t)(row0 + v) * ldc + col] = c[v];
}

// ---------- conversion kernels ----------------------------------------------

__global__ __launch_bounds__(256) void cvt_f32_bf16(const float* __restrict__ src,
                                                    unsigned short* __restrict__ dst,
                                                    int n) {
  for (int i = blockIdx.x * blockDim.x + threadIdx.x; i < n; i += gridDim.x * blockDim.x)
    dst[i] = f2bf(src[i]);
}

// Feature-reversed conversion: dst[r][c] = bf16(src[r][cols-1-c]).
// Implements torch.flip on the backward cell's input by flipping Wih_b0 columns.
__global__ __launch_bounds__(256) void cvt_rev_cols(const float* __restrict__ src,
                                                    unsigned short* __restrict__ dst,
                                                    int rows, int cols) {
  int n = rows * cols;
  for (int i = blockIdx.x * blockDim.x + threadIdx.x; i < n; i += gridDim.x * blockDim.x) {
    int r = i / cols, c = i - r * cols;
    dst[(size_t)r * cols + c] = f2bf(src[(size_t)r * cols + (cols - 1 - c)]);
  }
}

// ---------- batched input-projection GEMM (time-parallel) --------------------
// C[M,G] = A[M,K](bf16) * W[G,K](bf16)^T, fp32 out. 2x2 tiles per wave.
__global__ __launch_bounds__(256)
void gemm_bf16_nt(const unsigned short* __restrict__ A,
                  const unsigned short* __restrict__ W,
                  float* __restrict__ C,
                  int K, int ngrp /* G/32 = 48 */) {
  int wave = threadIdx.x >> 5;
  int lane = threadIdx.x & 31;
  int g  = blockIdx.x * 8 + wave;
  int mg = g / ngrp;
  int ng = g - mg * ngrp;

  const unsigned short* a0 = A + (size_t)(mg * 32) * K;
  const unsigned short* w0 = W + (size_t)(ng * 32) * K;

  v8f acc00 = {}, acc01 = {}, acc10 = {}, acc11 = {};
  for (int k0 = 0; k0 < K; k0 += 32) {
    __builtin_prefetch(w0 + k0 + 64, 0, 3);
    v16bf av0 = load_a_frag(a0 + k0, K, lane);
    v16bf av1 = load_a_frag(a0 + (size_t)16 * K + k0, K, lane);
    v16bf bv0 = load_b_frag(w0 + k0, K, lane);
    v16bf bv1 = load_b_frag(w0 + (size_t)16 * K + k0, K, lane);
    acc00 = __builtin_amdgcn_wmma_f32_16x16x32_bf16(false, av0, false, bv0, (short)0, acc00, false, false);
    acc01 = __builtin_amdgcn_wmma_f32_16x16x32_bf16(false, av0, false, bv1, (short)0, acc01, false, false);
    acc10 = __builtin_amdgcn_wmma_f32_16x16x32_bf16(false, av1, false, bv0, (short)0, acc10, false, false);
    acc11 = __builtin_amdgcn_wmma_f32_16x16x32_bf16(false, av1, false, bv1, (short)0, acc11, false, false);
  }
  float* c0 = C + (size_t)(mg * 32) * G_ + ng * 32;
  store_c(c0, G_, lane, acc00);
  store_c(c0 + 16, G_, lane, acc01);
  store_c(c0 + (size_t)16 * G_, G_, lane, acc10);
  store_c(c0 + (size_t)16 * G_ + 16, G_, lane, acc11);
}

// ---------- recurrent kernel -------------------------------------------------

struct RecurArgs {
  const float*          gi;       // [BT, G] fp32 precomputed x@Wih^T (no bias)
  const unsigned short* Whh;      // [G, H] bf16
  const float*          bih;      // [G]
  const float*          bhh;      // [G]
  unsigned short*       hseq;     // layer0: [BT, H] bf16 hidden sequence; else null
  float*                out;      // layer1: [B, T, 2H] fp32; else null
  float*                out_last; // layer1: [B, 2H]
  int                   dir;      // 0 = forward half, 1 = backward half of output
};

// One workgroup per GRU cell (blockIdx selects f/b). 32 waves = mt(0..3) x jg(0..7).
// Wave (mt,jg) owns output elements rows mt*16..+15, cols jg*64..+63 and computes
// the MATCHED gate-tile triples (jt, jt+32, jt+64) so r/z/n stay in registers.
// h (bf16, WMMA A operand) is double-buffered in LDS; fp32 h carry is in registers.
// Exactly one __syncthreads() per timestep.
__global__ __launch_bounds__(1024)
void gru_recurrence(RecurArgs a0, RecurArgs a1) {
  extern __shared__ unsigned short hbuf[];           // 2 * 64 * HS_
  RecurArgs a = (blockIdx.x == 0) ? a0 : a1;

  int tid = threadIdx.x;
  for (int i = tid; i < 2 * B_ * HS_; i += 1024) hbuf[i] = 0;   // h(-1) = 0
  __syncthreads();

  int lane = tid & 31, wave = tid >> 5;
  int mt = wave >> 3;                 // batch tile 0..3
  int jg = wave & 7;                  // hidden-column group 0..7 (4 j-tiles)
  int col  = lane & 15;
  int row0 = (lane & 16) ? 8 : 0;

  float hreg[4][8] = {};              // fp32 hidden carry, per-owned element
  float bir[4], bhr[4], biz[4], bhz[4], bin_[4], bhn[4];
#pragma unroll
  for (int ug = 0; ug < 4; ++ug) {
    int jcol = (jg * 4 + ug) * 16 + col;
    bir[ug]  = a.bih[jcol];            bhr[ug] = a.bhh[jcol];
    biz[ug]  = a.bih[H_ + jcol];       bhz[ug] = a.bhh[H_ + jcol];
    bin_[ug] = a.bih[2 * H_ + jcol];   bhn[ug] = a.bhh[2 * H_ + jcol];
  }

  for (int t = 0; t < T_; ++t) {
    const unsigned short* hrd = hbuf + (t & 1) * (B_ * HS_);
    unsigned short*       hwr = hbuf + ((t + 1) & 1) * (B_ * HS_);

#pragma unroll
    for (int ug = 0; ug < 4; ++ug) {
      int jt = jg * 4 + ug;
      const unsigned short* wr = a.Whh + (size_t)(jt * 16) * H_;          // r rows
      const unsigned short* wz = a.Whh + (size_t)((32 + jt) * 16) * H_;   // z rows
      const unsigned short* wn = a.Whh + (size_t)((64 + jt) * 16) * H_;   // n rows

      v8f ar = {}, az = {}, an = {};
      for (int k0 = 0; k0 < H_; k0 += 32) {
        __builtin_prefetch(wr + k0 + 64, 0, 3);
        v16bf av = load_a_frag_lds(hrd + (mt * 16) * HS_ + k0, HS_, lane);
        v16bf br = load_b_frag(wr + k0, H_, lane);
        v16bf bz = load_b_frag(wz + k0, H_, lane);
        v16bf bn = load_b_frag(wn + k0, H_, lane);
        ar = __builtin_amdgcn_wmma_f32_16x16x32_bf16(false, av, false, br, (short)0, ar, false, false);
        az = __builtin_amdgcn_wmma_f32_16x16x32_bf16(false, av, false, bz, (short)0, az, false, false);
        an = __builtin_amdgcn_wmma_f32_16x16x32_bf16(false, av, false, bn, (short)0, an, false, false);
      }

      // gates in registers (fp32), PyTorch GRUCell math
      int jcol = jt * 16 + col;
#pragma unroll
      for (int v = 0; v < 8; ++v) {
        int bb = mt * 16 + row0 + v;                       // batch index
        size_t grow = ((size_t)bb * T_ + t) * G_;
        float xr = a.gi[grow + jcol]          + bir[ug]  + ar[v] + bhr[ug];
        float xz = a.gi[grow + H_ + jcol]     + biz[ug]  + az[v] + bhz[ug];
        float r  = 1.0f / (1.0f + __expf(-xr));
        float z  = 1.0f / (1.0f + __expf(-xz));
        float pre = a.gi[grow + 2 * H_ + jcol] + bin_[ug] + r * (an[v] + bhn[ug]);
        float e2  = __expf(-2.0f * pre);
        float nn  = (1.0f - e2) / (1.0f + e2);             // tanh(pre)
        float hnew = (1.0f - z) * nn + z * hreg[ug][v];
        hreg[ug][v] = hnew;

        hwr[bb * HS_ + jcol] = f2bf(hnew);                 // next step's A operand
        if (a.hseq)
          a.hseq[((size_t)bb * T_ + t) * H_ + jcol] = f2bf(hnew);
        if (a.out) {
          a.out[((size_t)bb * T_ + t) * (2 * H_) + a.dir * H_ + jcol] = hnew;
          if (t == T_ - 1)
            a.out_last[(size_t)bb * (2 * H_) + a.dir * H_ + jcol] = hnew;
        }
      }
    }
    __syncthreads();   // all hwr writes visible before anyone reads it at t+1
  }
}

// ---------- host launcher ----------------------------------------------------

extern "C" void kernel_launch(void* const* d_in, const int* in_sizes, int n_in,
                              void* d_out, int out_size, void* d_ws, size_t ws_size,
                              hipStream_t stream) {
  (void)in_sizes; (void)n_in; (void)out_size; (void)ws_size;

  const float* x      = (const float*)d_in[0];
  const float* Wih_f0 = (const float*)d_in[1];
  const float* Whh_f0 = (const float*)d_in[2];
  const float* bih_f0 = (const float*)d_in[3];
  const float* bhh_f0 = (const float*)d_in[4];
  const float* Wih_f1 = (const float*)d_in[5];
  const float* Whh_f1 = (const float*)d_in[6];
  const float* bih_f1 = (const float*)d_in[7];
  const float* bhh_f1 = (const float*)d_in[8];
  const float* Wih_b0 = (const float*)d_in[9];
  const float* Whh_b0 = (const float*)d_in[10];
  const float* bih_b0 = (const float*)d_in[11];
  const float* bhh_b0 = (const float*)d_in[12];
  const float* Wih_b1 = (const float*)d_in[13];
  const float* Whh_b1 = (const float*)d_in[14];
  const float* bih_b1 = (const float*)d_in[15];
  const float* bhh_b1 = (const float*)d_in[16];

  // workspace carve-out (~497 MB total)
  char* ws = (char*)d_ws;
  size_t off = 0;
  auto carve = [&](size_t bytes) -> char* {
    char* p = ws + off;
    off += (bytes + 255) & ~(size_t)255;
    return p;
  };
  unsigned short* xb     = (unsigned short*)carve((size_t)BT_ * I_ * 2);
  unsigned short* wihf0b = (unsigned short*)carve((size_t)G_ * I_ * 2);
  unsigned short* wihb0b = (unsigned short*)carve((size_t)G_ * I_ * 2);  // column-reversed
  unsigned short* whhf0b = (unsigned short*)carve((size_t)G_ * H_ * 2);
  unsigned short* whhb0b = (unsigned short*)carve((size_t)G_ * H_ * 2);
  unsigned short* wihf1b = (unsigned short*)carve((size_t)G_ * H_ * 2);
  unsigned short* wihb1b = (unsigned short*)carve((size_t)G_ * H_ * 2);
  unsigned short* whhf1b = (unsigned short*)carve((size_t)G_ * H_ * 2);
  unsigned short* whhb1b = (unsigned short*)carve((size_t)G_ * H_ * 2);
  float*          gif    = (float*)carve((size_t)BT_ * G_ * 4);
  float*          gib    = (float*)carve((size_t)BT_ * G_ * 4);
  unsigned short* hseqf  = (unsigned short*)carve((size_t)BT_ * H_ * 2);
  unsigned short* hseqb  = (unsigned short*)carve((size_t)BT_ * H_ * 2);

  // 1) bf16 conversions (x and weights)
  cvt_f32_bf16<<<4096, 256, 0, stream>>>(x,      xb,     BT_ * I_);
  cvt_f32_bf16<<<1536, 256, 0, stream>>>(Wih_f0, wihf0b, G_ * I_);
  cvt_rev_cols<<<1536, 256, 0, stream>>>(Wih_b0, wihb0b, G_, I_);   // feature flip folded in
  cvt_f32_bf16<<<3072, 256, 0, stream>>>(Whh_f0, whhf0b, G_ * H_);
  cvt_f32_bf16<<<3072, 256, 0, stream>>>(Whh_b0, whhb0b, G_ * H_);
  cvt_f32_bf16<<<3072, 256, 0, stream>>>(Wih_f1, wihf1b, G_ * H_);
  cvt_f32_bf16<<<3072, 256, 0, stream>>>(Wih_b1, wihb1b, G_ * H_);
  cvt_f32_bf16<<<3072, 256, 0, stream>>>(Whh_f1, whhf1b, G_ * H_);
  cvt_f32_bf16<<<3072, 256, 0, stream>>>(Whh_b1, whhb1b, G_ * H_);

  const int ngrp   = G_ / 32;                    // 48 (2-tile groups in N)
  const int blocks = (BT_ / 32) * ngrp / 8;      // 6144 blocks, 8 waves each
  const size_t lds_bytes = (size_t)2 * B_ * HS_ * 2;  // 133,120 B (< 320 KB WGP LDS)

  // 2) layer-0 input projections for ALL timesteps (time-parallel)
  gemm_bf16_nt<<<blocks, 256, 0, stream>>>(xb, wihf0b, gif, I_, ngrp);
  gemm_bf16_nt<<<blocks, 256, 0, stream>>>(xb, wihb0b, gib, I_, ngrp);

  // 3) layer-0 recurrences (forward + backward cells concurrent)
  RecurArgs f0{gif, whhf0b, bih_f0, bhh_f0, hseqf, nullptr, nullptr, 0};
  RecurArgs b0{gib, whhb0b, bih_b0, bhh_b0, hseqb, nullptr, nullptr, 0};
  gru_recurrence<<<2, 1024, lds_bytes, stream>>>(f0, b0);

  // 4) layer-1 input projections (reuse gi buffers)
  gemm_bf16_nt<<<blocks, 256, 0, stream>>>(hseqf, wihf1b, gif, H_, ngrp);
  gemm_bf16_nt<<<blocks, 256, 0, stream>>>(hseqb, wihb1b, gib, H_, ngrp);

  // 5) layer-1 recurrences -> final outputs
  float* out      = (float*)d_out;
  float* out_last = out + (size_t)B_ * T_ * 2 * H_;
  RecurArgs f1{gif, whhf1b, bih_f1, bhh_f1, nullptr, out, out_last, 0};
  RecurArgs b1{gib, whhb1b, bih_b1, bhh_b1, nullptr, out, out_last, 1};
  gru_recurrence<<<2, 1024, lds_bytes, stream>>>(f1, b1);
}
